// Retriever_73212012527961
// MI455X (gfx1250) — matrix operand order repaired and checked by reading, not compiled, not run on hardware
//
#include <hip/hip_runtime.h>
#include <hip/hip_bf16.h>
#include <math.h>
#include <stdint.h>

#define DEV static __device__ __forceinline__

typedef __attribute__((ext_vector_type(8)))  float          v8f;
typedef __attribute__((ext_vector_type(16))) __bf16         v16bf;
typedef __attribute__((ext_vector_type(8)))  unsigned short ushort8;
typedef __attribute__((ext_vector_type(16))) unsigned short ushort16;
typedef __attribute__((ext_vector_type(4)))  unsigned int   uint4v;
typedef __attribute__((ext_vector_type(8)))  int            int8v;
typedef __attribute__((ext_vector_type(4)))  int            int4v;

// Tensor Data Mover availability. This toolchain (clang-23 / therock-10.0 headers)
// exposes the 6-arg builtin: (uint32x4, int32x8, int32x4, int32x4, int32x8, i32 cpol).
#if defined(__has_builtin)
#if __has_builtin(__builtin_amdgcn_tensor_load_to_lds)
#define USE_TDM 1
#endif
#endif
#ifndef USE_TDM
#define USE_TDM 0
#endif

// ---------------- constants (match reference) ----------------
static constexpr int B_ = 8, S_ = 256, D_ = 768, H_ = 12, L_ = 12;
static constexpr int N_ = 100000, DEG_ = 16;
static constexpr int TOPW = 50;        // NUM_WALKS (only first 50 of top_k used)
static constexpr int WALKLEN = 199;
static constexpr int ROWS = B_ * S_;   // 2048
static constexpr float THRESH = 0.75f;

// ---------------- bf16 helpers ----------------
DEV unsigned short f2bfu(float f) {
  unsigned u = __builtin_bit_cast(unsigned, f);
  u += 0x7fffu + ((u >> 16) & 1u);        // round-to-nearest-even
  return (unsigned short)(u >> 16);
}
DEV __bf16 f2bf(float f) {
  unsigned short s = f2bfu(f);
  return __builtin_bit_cast(__bf16, s);
}
// Two 16-byte LDS loads -> one 16-element bf16 fragment.
DEV v16bf frag_ld(const unsigned short* plo, const unsigned short* phi) {
  ushort8 lo = *(const ushort8*)plo;
  ushort8 hi = *(const ushort8*)phi;
  ushort16 f = __builtin_shufflevector(lo, hi, 0,1,2,3,4,5,6,7,8,9,10,11,12,13,14,15);
  return __builtin_bit_cast(v16bf, f);
}
// B operand loaded K-sequential per lane, same order as its A (lanes 0-15 K=0..15, lanes 16-31 K=16..31):
// one contiguous 32-byte run.
DEV v16bf frag_seq(const unsigned short* p) { return frag_ld(p, p + 8); }

DEV v8f wmma_bf16(v16bf a, v16bf b, v8f c) {
  // D = A(16x32 bf16) x B(32x16 bf16) + C(16x16 f32)
  return __builtin_amdgcn_wmma_f32_16x16x32_bf16(false, a, false, b, (short)0, c, false, false);
}

DEV void wait_tensor0() {
#if defined(__has_builtin) && __has_builtin(__builtin_amdgcn_s_wait_tensorcnt)
  __builtin_amdgcn_s_wait_tensorcnt(0);
#elif defined(__gfx1250__)
  asm volatile("s_wait_tensorcnt 0x0" ::: "memory");
#endif
}

// ---------------- TDM 2-D tile load: global f32 tile -> LDS ----------------
// D# group0/group1 per CDNA5 ISA (2-D, data_size=4B).
DEV void tdm_load_tile_f32(float* ldsTile, const float* gsrc,
                           unsigned tileX, unsigned tileY,
                           unsigned tensorDim0, unsigned tensorDim1,
                           unsigned stride0elems) {
#if USE_TDM
  unsigned lds = (unsigned)(unsigned long long)(uintptr_t)ldsTile; // low 32 bits = LDS offset
  unsigned long long ga = (unsigned long long)(uintptr_t)gsrc;
  uint4v g0;
  g0[0] = 1u;                                          // count=1 (valid), no gather
  g0[1] = lds;                                         // lds_addr
  g0[2] = (unsigned)(ga & 0xffffffffu);                // global_addr[31:0]
  g0[3] = (unsigned)((ga >> 32) & 0x01ffffffu) | (2u << 30); // addr[56:32] | type=2
  int8v g1;
  g1[0] = (int)(2u << 16);                             // workgroup_mask=0, data_size=2 (4B)
  g1[1] = (int)((tensorDim0 & 0xffffu) << 16);         // tensor_dim0[15:0] @ bits63:48
  g1[2] = (int)((tensorDim0 >> 16) & 0xffffu) |        // tensor_dim0[31:16]
          (int)((tensorDim1 & 0xffffu) << 16);         // tensor_dim1[15:0]
  g1[3] = (int)((tensorDim1 >> 16) & 0xffffu) |        // tensor_dim1[31:16]
          (int)((tileX & 0xffffu) << 16);              // tile_dim0
  g1[4] = (int)(tileY & 0xffffu);                      // tile_dim1 (tile_dim2=0 -> 2D)
  g1[5] = (int)stride0elems;                           // tensor_dim0_stride[31:0]
  g1[6] = 0;                                           // stride[47:32]=0, dim1_stride lo=0
  g1[7] = 0;
  int4v z4 = {};
  int8v z8 = {};
  __builtin_amdgcn_tensor_load_to_lds(g0, g1, z4, z4, z8, 0);
#else
  (void)ldsTile; (void)gsrc; (void)tileX; (void)tileY;
  (void)tensorDim0; (void)tensorDim1; (void)stride0elems;
#endif
}

// ---------------- block reduction (wave32) ----------------
DEV float block_sum(float v, float* red) {
#pragma unroll
  for (int o = 16; o > 0; o >>= 1) v += __shfl_xor(v, o, 32);
  int w = threadIdx.x >> 5;
  if ((threadIdx.x & 31) == 0) red[w] = v;
  __syncthreads();
  if (threadIdx.x < 8) {
    float r = red[threadIdx.x];
#pragma unroll
    for (int o = 4; o > 0; o >>= 1) r += __shfl_xor(r, o, 32);
    if (threadIdx.x == 0) red[0] = r;
  }
  __syncthreads();
  float out = red[0];
  __syncthreads();
  return out;
}

// ---------------- embedding gather ----------------
__global__ __launch_bounds__(256) void embed_kernel(
    const int* __restrict__ ids, const float* __restrict__ tok,
    const float* __restrict__ pos, float* __restrict__ x) {
  int row = blockIdx.x;               // b*S + s
  int s = row & (S_ - 1);
  int id = ids[row];
  const float* tr = tok + (size_t)id * D_;
  const float* pr = pos + (size_t)s * D_;
  float* xr = x + (size_t)row * D_;
#pragma unroll
  for (int j = 0; j < 3; ++j) {
    int k = threadIdx.x + 256 * j;
    xr[k] = tr[k] + pr[k];
  }
}

// ---------------- LayerNorm (in==out allowed) ----------------
__global__ __launch_bounds__(256) void ln_kernel(
    const float* __restrict__ in, const float* __restrict__ g,
    const float* __restrict__ bb, float* __restrict__ out) {
  __shared__ float red[8];
  int row = blockIdx.x, t = threadIdx.x;
  const float* xr = in + (size_t)row * D_;
  float v0 = xr[t], v1 = xr[t + 256], v2 = xr[t + 512];
  float m = block_sum(v0 + v1 + v2, red) * (1.0f / D_);
  float d0 = v0 - m, d1 = v1 - m, d2 = v2 - m;
  float var = block_sum(d0 * d0 + d1 * d1 + d2 * d2, red) * (1.0f / D_);
  float inv = rsqrtf(var + 1e-12f);
  float* o = out + (size_t)row * D_;
  o[t]       = d0 * inv * g[t]       + bb[t];
  o[t + 256] = d1 * inv * g[t + 256] + bb[t + 256];
  o[t + 512] = d2 * inv * g[t + 512] + bb[t + 512];
}

// ---------------- generic WMMA GEMM: out = act(A@B + bias [+ residual]) ----------------
// A: MxK f32 row-major, B: KxN f32 row-major. M%64==0, N%64==0, K%32==0.
// 64x64 tile / 256-thread block (8 waves x 2 wmma tiles), double-buffered LDS.
// B tile staged by the Tensor Data Mover (f32), converted to bf16 at fragment build.
__global__ __launch_bounds__(256) void gemm_kernel(
    const float* __restrict__ A, const float* __restrict__ B,
    const float* __restrict__ bias, const float* __restrict__ residual,
    float* __restrict__ out, int M, int N, int K, int act) {
  (void)M;
  __shared__ __attribute__((aligned(16))) unsigned short As[2][64 * 40]; // bf16 [m][k]
  __shared__ __attribute__((aligned(16))) float          Bs[2][32 * 64]; // f32  [k][n]
  int m0 = blockIdx.y * 64, n0 = blockIdx.x * 64;
  int t = threadIdx.x, lane = t & 31, w = t >> 5;
  int mt = w & 3, ntb = (w >> 2) * 2;
  int kh = lane >> 4, kh8 = kh * 8;
  v8f cc0 = {}, cc1 = {};

  // ---- staging helpers (lambdas keep LDS arrays in scope) ----
  auto stageA = [&](int k0, unsigned short* dst) {
    int r = t >> 2, kb = (t & 3) * 8;                 // 64 rows x 32 k, 8 elems/thread
    const float* src = &A[(size_t)(m0 + r) * K + k0 + kb];
    float4 f0 = *(const float4*)src;
    float4 f1 = *(const float4*)(src + 4);
    ushort8 u;
    u[0] = f2bfu(f0.x); u[1] = f2bfu(f0.y); u[2] = f2bfu(f0.z); u[3] = f2bfu(f0.w);
    u[4] = f2bfu(f1.x); u[5] = f2bfu(f1.y); u[6] = f2bfu(f1.z); u[7] = f2bfu(f1.w);
    *(ushort8*)&dst[r * 40 + kb] = u;
  };
  auto stageB = [&](int k0, float* dst) {
#if USE_TDM
    if (w == 0)
      tdm_load_tile_f32(dst, B + (size_t)k0 * N + n0,
                        /*tileX=*/64, /*tileY=*/32,
                        /*tensorDim0=*/(unsigned)N, /*tensorDim1=*/(unsigned)K,
                        /*stride0=*/(unsigned)N);
#else
    int kk = t >> 3, nb = (t & 7) * 8;                // 32 k x 64 n, 8 elems/thread
    const float* src = &B[(size_t)(k0 + kk) * N + n0 + nb];
    float4 f0 = *(const float4*)src, f1 = *(const float4*)(src + 4);
    *(float4*)&dst[kk * 64 + nb] = f0;
    *(float4*)&dst[kk * 64 + nb + 4] = f1;
#endif
  };
  auto fragB = [&](const float* bs, int n) {          // B fragment in A's k order (8h+i, then 16+8h+i-8), f32->bf16
    v16bf b;
#pragma unroll
    for (int i = 0; i < 16; ++i) b[i] = f2bf(bs[((i < 8 ? i : i + 8) + kh8) * 64 + n]);
    return b;
  };

  // prologue: stage chunk 0 into buffer 0
  stageB(0, Bs[0]);
  stageA(0, As[0]);
  int nchunks = K / 32;
  for (int ci = 0; ci < nchunks; ++ci) {
    int cur = ci & 1, nxt = cur ^ 1;
#if USE_TDM
    if (w == 0) wait_tensor0();                       // TDM for Bs[cur] complete
#endif
    __syncthreads();                                  // staging for buffer `cur` visible
    if (ci + 1 < nchunks) {                           // overlap next-chunk staging
      stageB((ci + 1) * 32, Bs[nxt]);
      stageA((ci + 1) * 32, As[nxt]);
      __builtin_prefetch(&A[(size_t)(m0 + (t >> 2)) * K + (ci + 1) * 32 + (t & 3) * 8], 0, 1);
    }
    int ar = (mt * 16 + (lane & 15)) * 40;
    v16bf a  = frag_ld(&As[cur][ar + kh8], &As[cur][ar + 16 + kh8]);
    v16bf b0 = fragB(Bs[cur], (ntb    ) * 16 + (lane & 15));
    v16bf b1 = fragB(Bs[cur], (ntb + 1) * 16 + (lane & 15));
    cc0 = wmma_bf16(a, b0, cc0);
    cc1 = wmma_bf16(a, b1, cc1);
  }
  // epilogue; C layout: lane -> n = l&15, VGPR j -> m = j + 8*(l>>4)
#pragma unroll
  for (int j = 0; j < 2; ++j) {
    v8f c = j ? cc1 : cc0;
    int col = n0 + (ntb + j) * 16 + (lane & 15);
    float bv = bias[col];
#pragma unroll
    for (int v = 0; v < 8; ++v) {
      int mrow = m0 + mt * 16 + v + 8 * kh;
      size_t off = (size_t)mrow * N + col;
      float val = c[v] + bv;
      if (residual) val += residual[off];
      if (act == 1) val = 0.5f * val * (1.0f + erff(val * 0.70710678118654752f));
      out[off] = val;
    }
  }
}

// ---------------- attention: per (b, h, 32-query tile) ----------------
__global__ __launch_bounds__(256) void attn_kernel(
    const float* __restrict__ qkv, float* __restrict__ o) {
  __shared__ __attribute__((aligned(16))) unsigned short Qs[32 * 72]; // [q][dh] bf16
  __shared__ __attribute__((aligned(16))) unsigned short Ks[64 * 72]; // [key][dh] bf16
  __shared__ __attribute__((aligned(16))) unsigned short Vt[64 * 40]; // [dh][key] bf16
  __shared__ float Pf[32 * 264];                                      // scores / probs
  int qt = blockIdx.x & 7;
  int h  = (blockIdx.x >> 3) % H_;
  int b  = blockIdx.x / (8 * H_);
  int t = threadIdx.x, lane = t & 31, w = t >> 5;
  int kh = lane >> 4, kh8 = kh * 8;
  const size_t RS = 3 * D_;   // qkv row stride 2304
  const float scale = 0.125f; // 1/sqrt(64)

  {                                                   // stage Q 32x64 (vectorized)
    int r = t >> 3, cb = (t & 7) * 8;
    const float* src = &qkv[(size_t)(b * S_ + qt * 32 + r) * RS + h * 64 + cb];
    float4 f0 = *(const float4*)src, f1 = *(const float4*)(src + 4);
    ushort8 u;
    u[0] = f2bfu(f0.x); u[1] = f2bfu(f0.y); u[2] = f2bfu(f0.z); u[3] = f2bfu(f0.w);
    u[4] = f2bfu(f1.x); u[5] = f2bfu(f1.y); u[6] = f2bfu(f1.z); u[7] = f2bfu(f1.w);
    *(ushort8*)&Qs[r * 72 + cb] = u;
  }
  // ---- scores: S = (Q K^T) * scale, 4 key chunks of 64 ----
  for (int kc = 0; kc < 4; ++kc) {
    __syncthreads();
    {                                                 // stage K 64x64 (vectorized)
      int r = t >> 2, cb = (t & 3) * 16;
      const float* src = &qkv[(size_t)(b * S_ + kc * 64 + r) * RS + D_ + h * 64 + cb];
#pragma unroll
      for (int half = 0; half < 2; ++half) {
        float4 f0 = *(const float4*)(src + half * 8), f1 = *(const float4*)(src + half * 8 + 4);
        ushort8 u;
        u[0] = f2bfu(f0.x); u[1] = f2bfu(f0.y); u[2] = f2bfu(f0.z); u[3] = f2bfu(f0.w);
        u[4] = f2bfu(f1.x); u[5] = f2bfu(f1.y); u[6] = f2bfu(f1.z); u[7] = f2bfu(f1.w);
        *(ushort8*)&Ks[r * 72 + cb + half * 8] = u;
      }
    }
    __syncthreads();
    int mt = w & 1, nt = w >> 1;                      // 2 x 4 tiles, one per wave
    v8f c = {};
#pragma unroll
    for (int kk = 0; kk < 2; ++kk) {                  // DH=64 -> two 32-chunks
      int ar = (mt * 16 + (lane & 15)) * 72 + kk * 32;
      v16bf a  = frag_ld(&Qs[ar + 16 * kh], &Qs[ar + 16 * kh + 8]); // A: sequential K
      v16bf bb = frag_seq(&Ks[(nt * 16 + (lane & 15)) * 72 + kk * 32 + 16 * kh]); // B: sequential K
      c = wmma_bf16(a, bb, c);
    }
#pragma unroll
    for (int v = 0; v < 8; ++v) {
      int ml = mt * 16 + v + 8 * kh;
      int nl = kc * 64 + nt * 16 + (lane & 15);
      Pf[ml * 264 + nl] = c[v] * scale;
    }
  }
  __syncthreads();
  // ---- softmax rows (exact: sub max, expf, normalize) ----
  if (t < 32) {
    float mx = -1e30f;
    for (int i = 0; i < S_; ++i) mx = fmaxf(mx, Pf[t * 264 + i]);
    float sum = 0.f;
    for (int i = 0; i < S_; ++i) { float e = expf(Pf[t * 264 + i] - mx); Pf[t * 264 + i] = e; sum += e; }
    float inv = 1.0f / sum;
    for (int i = 0; i < S_; ++i) Pf[t * 264 + i] *= inv;
  }
  // ---- O = P @ V, K=256 in 8 chunks of 32 ----
  int mt = w & 1, nt = w >> 1;
  v8f c = {};
  for (int kc = 0; kc < 8; ++kc) {
    __syncthreads();
#pragma unroll
    for (int j = 0; j < 8; ++j) {                     // stage V chunk 32(key) x 64(dh), transposed
      int e = t + 256 * j, kk = e >> 6, d = e & 63;
      Vt[d * 40 + kk] = f2bfu(qkv[(size_t)(b * S_ + kc * 32 + kk) * RS + 2 * D_ + h * 64 + d]);
    }
    __syncthreads();
    v16bf a;
    int m = mt * 16 + (lane & 15);
#pragma unroll
    for (int i = 0; i < 16; ++i) {                    // A frag from f32 probs (sequential K)
      int k = i + 16 * kh;
      a[i] = f2bf(Pf[m * 264 + kc * 32 + k]);
    }
    v16bf bb = frag_seq(&Vt[(nt * 16 + (lane & 15)) * 40 + 16 * kh]); // B: sequential K
    c = wmma_bf16(a, bb, c);
  }
#pragma unroll
  for (int v = 0; v < 8; ++v) {
    int ml = mt * 16 + v + 8 * kh;
    int nl = nt * 16 + (lane & 15);
    o[(size_t)(b * S_ + qt * 32 + ml) * D_ + h * 64 + nl] = c[v];
  }
}

// ---------------- pooled = tanh(x[:,0] @ Wp + bp); also 1/||pooled|| ----------------
__global__ __launch_bounds__(256) void pooled_kernel(
    const float* __restrict__ x, const float* __restrict__ Wp, const float* __restrict__ bp,
    float* __restrict__ pooled, float* __restrict__ pninv, float* __restrict__ outq) {
  __shared__ float xs[D_];
  __shared__ float red[8];
  int b = blockIdx.x, t = threadIdx.x;
  const float* xr = x + (size_t)(b * S_) * D_;    // CLS row
  xs[t] = xr[t]; xs[t + 256] = xr[t + 256]; xs[t + 512] = xr[t + 512];
  __syncthreads();
  float ss = 0.f;
#pragma unroll
  for (int j = 0; j < 3; ++j) {
    int n = t + 256 * j;
    float acc = bp[n];
    for (int k = 0; k < D_; ++k) acc = fmaf(xs[k], Wp[(size_t)k * D_ + n], acc);
    float p = tanhf(acc);
    pooled[b * D_ + n] = p;
    outq[(size_t)b * (2 * D_) + n] = p;           // first half of qwp
    ss += p * p;
  }
  float s = block_sum(ss, red);
  if (t == 0) pninv[b] = rsqrtf(s);
}

// ---------------- cosine sim: one wave per index row ----------------
__global__ __launch_bounds__(256) void sim_kernel(
    const float* __restrict__ emb, const float* __restrict__ pooled,
    const float* __restrict__ pninv, float* __restrict__ sim) {
  __shared__ float Pl[B_ * D_];
  __shared__ float Pn[B_];
  int t = threadIdx.x;
#pragma unroll
  for (int j = 0; j < 24; ++j) Pl[t + 256 * j] = pooled[t + 256 * j];
  if (t < B_) Pn[t] = pninv[t];
  __syncthreads();
  int w = t >> 5, lane = t & 31;
  int i = blockIdx.x * 8 + w;                     // 12500*8 == N
  float nn = 0.f, d[B_];
#pragma unroll
  for (int b = 0; b < B_; ++b) d[b] = 0.f;
  for (int c = 0; c < 24; ++c) {
    int k = lane + 32 * c;
    float e = emb[(size_t)i * D_ + k];
    nn = fmaf(e, e, nn);
#pragma unroll
    for (int b = 0; b < B_; ++b) d[b] = fmaf(e, Pl[b * D_ + k], d[b]);
  }
#pragma unroll
  for (int o = 16; o > 0; o >>= 1) {
    nn += __shfl_xor(nn, o, 32);
#pragma unroll
    for (int b = 0; b < B_; ++b) d[b] += __shfl_xor(d[b], o, 32);
  }
  if (lane == 0) {
    float inv = rsqrtf(nn);
#pragma unroll
    for (int b = 0; b < B_; ++b) sim[(size_t)b * N_ + i] = d[b] * inv * Pn[b];
  }
}

// ---------------- top-50 per batch (value desc, index asc), mutation-free ----------------
__global__ __launch_bounds__(256) void topk_kernel(
    const float* __restrict__ sim, int* __restrict__ seeds) {
  __shared__ float rv[256];
  __shared__ int ri[256];
  __shared__ float bvs;
  __shared__ int bis;
  int b = blockIdx.x, t = threadIdx.x;
  const float* s = sim + (size_t)b * N_;
  float pv = 1e30f; int pi = -1;
  for (int sel = 0; sel < TOPW; ++sel) {
    float v = -1e30f; int id = 0x7fffffff;
    for (int i = t; i < N_; i += 256) {
      float sv = s[i];
      bool elig = (sv < pv) || (sv == pv && i > pi);
      if (elig && (sv > v || (sv == v && i < id))) { v = sv; id = i; }
    }
    rv[t] = v; ri[t] = id;
    __syncthreads();
    for (int st = 128; st > 0; st >>= 1) {
      if (t < st) {
        float ov = rv[t + st]; int oi = ri[t + st];
        if (ov > rv[t] || (ov == rv[t] && oi < ri[t])) { rv[t] = ov; ri[t] = oi; }
      }
      __syncthreads();
    }
    if (t == 0) { seeds[b * TOPW + sel] = ri[0]; bvs = rv[0]; bis = ri[0]; }
    __syncthreads();
    pv = bvs; pi = bis;
  }
}

__global__ __launch_bounds__(256) void zero_kernel(float* __restrict__ p, int n) {
  int i = blockIdx.x * 256 + threadIdx.x;
  if (i < n) p[i] = 0.f;
}

// ---------------- random walks: one thread per walk ----------------
__global__ __launch_bounds__(256) void walk_kernel(
    const int* __restrict__ seeds, const int* __restrict__ neigh,
    const int* __restrict__ wr, float* __restrict__ counts) {
  int t = blockIdx.x * 256 + threadIdx.x;
  if (t >= B_ * TOPW) return;
  int b = t / TOPW, w = t % TOPW;
  int cur = seeds[b * TOPW + w];
  atomicAdd(&counts[(size_t)b * N_ + cur], 1.0f);
  const int* r = wr + (size_t)(b * TOPW + w) * WALKLEN;
  for (int s = 0; s < WALKLEN; ++s) {
    cur = neigh[(size_t)cur * DEG_ + r[s]];
    atomicAdd(&counts[(size_t)b * N_ + cur], 1.0f);
  }
}

// ---------------- best passage: argmax(where(sim>=0.75, counts, -1)), gather emb ----------------
__global__ __launch_bounds__(256) void best_kernel(
    const float* __restrict__ sim, const float* __restrict__ counts,
    const float* __restrict__ emb, float* __restrict__ outq) {
  __shared__ float rv[256];
  __shared__ int ri[256];
  __shared__ int bidx;
  int b = blockIdx.x, t = threadIdx.x;
  float v = -1e30f; int id = 0x7fffffff;
  for (int i = t; i < N_; i += 256) {
    float sc = (sim[(size_t)b * N_ + i] >= THRESH) ? counts[(size_t)b * N_ + i] : -1.0f;
    if (sc > v || (sc == v && i < id)) { v = sc; id = i; }
  }
  rv[t] = v; ri[t] = id;
  __syncthreads();
  for (int st = 128; st > 0; st >>= 1) {
    if (t < st) {
      float ov = rv[t + st]; int oi = ri[t + st];
      if (ov > rv[t] || (ov == rv[t] && oi < ri[t])) { rv[t] = ov; ri[t] = oi; }
    }
    __syncthreads();
  }
  if (t == 0) bidx = ri[0];
  __syncthreads();
#pragma unroll
  for (int j = 0; j < 3; ++j) {
    int k = t + 256 * j;
    outq[(size_t)b * (2 * D_) + D_ + k] = emb[(size_t)bidx * D_ + k];
  }
}

// ---------------- host launch ----------------
extern "C" void kernel_launch(void* const* d_in, const int* in_sizes, int n_in,
                              void* d_out, int out_size, void* d_ws, size_t ws_size,
                              hipStream_t stream) {
  (void)in_sizes; (void)n_in; (void)out_size; (void)ws_size;
  const int*   input_ids = (const int*)d_in[0];
  const int*   walk_rand = (const int*)d_in[1];
  const int*   neighbors = (const int*)d_in[2];
  const float* tok_emb   = (const float*)d_in[3];
  const float* pos_emb   = (const float*)d_in[4];
  const float* emb_g     = (const float*)d_in[5];
  const float* emb_b     = (const float*)d_in[6];
  const float* Wqkv      = (const float*)d_in[7];
  const float* bqkv      = (const float*)d_in[8];
  const float* Wo        = (const float*)d_in[9];
  const float* bo        = (const float*)d_in[10];
  const float* ln1_g     = (const float*)d_in[11];
  const float* ln1_b     = (const float*)d_in[12];
  const float* Wff1      = (const float*)d_in[13];
  const float* bff1      = (const float*)d_in[14];
  const float* Wff2      = (const float*)d_in[15];
  const float* bff2      = (const float*)d_in[16];
  const float* ln2_g     = (const float*)d_in[17];
  const float* ln2_b     = (const float*)d_in[18];
  const float* Wp        = (const float*)d_in[19];
  const float* bp        = (const float*)d_in[20];
  const float* index_emb = (const float*)d_in[21];
  float* out = (float*)d_out;

  float* ws = (float*)d_ws;
  size_t off = 0;
  float* x      = ws + off; off += (size_t)ROWS * D_;        // 2048x768
  float* qkv    = ws + off; off += (size_t)ROWS * 3 * D_;    // 2048x2304
  float* attn   = ws + off; off += (size_t)ROWS * D_;
  float* tmp    = ws + off; off += (size_t)ROWS * D_;
  float* ffh    = ws + off; off += (size_t)ROWS * 4 * D_;    // 2048x3072
  float* sim    = ws + off; off += (size_t)B_ * N_;
  float* counts = ws + off; off += (size_t)B_ * N_;
  float* pooled = ws + off; off += (size_t)B_ * D_;
  float* pninv  = ws + off; off += B_;
  int*   seeds  = (int*)(ws + off); off += B_ * TOPW;

  // ---- encoder ----
  embed_kernel<<<ROWS, 256, 0, stream>>>(input_ids, tok_emb, pos_emb, x);
  ln_kernel<<<ROWS, 256, 0, stream>>>(x, emb_g, emb_b, x);
  for (int l = 0; l < L_; ++l) {
    gemm_kernel<<<dim3(3 * D_ / 64, ROWS / 64), 256, 0, stream>>>(
        x, Wqkv + (size_t)l * D_ * 3 * D_, bqkv + (size_t)l * 3 * D_, nullptr,
        qkv, ROWS, 3 * D_, D_, 0);
    attn_kernel<<<B_ * H_ * (S_ / 32), 256, 0, stream>>>(qkv, attn);
    gemm_kernel<<<dim3(D_ / 64, ROWS / 64), 256, 0, stream>>>(
        attn, Wo + (size_t)l * D_ * D_, bo + (size_t)l * D_, x,
        tmp, ROWS, D_, D_, 0);
    ln_kernel<<<ROWS, 256, 0, stream>>>(tmp, ln1_g + (size_t)l * D_, ln1_b + (size_t)l * D_, x);
    gemm_kernel<<<dim3(4 * D_ / 64, ROWS / 64), 256, 0, stream>>>(
        x, Wff1 + (size_t)l * D_ * 4 * D_, bff1 + (size_t)l * 4 * D_, nullptr,
        ffh, ROWS, 4 * D_, D_, 1);
    gemm_kernel<<<dim3(D_ / 64, ROWS / 64), 256, 0, stream>>>(
        ffh, Wff2 + (size_t)l * 4 * D_ * D_, bff2 + (size_t)l * D_, x,
        tmp, ROWS, D_, 4 * D_, 0);
    ln_kernel<<<ROWS, 256, 0, stream>>>(tmp, ln2_g + (size_t)l * D_, ln2_b + (size_t)l * D_, x);
  }
  // sent output
  (void)hipMemcpyAsync(out, x, sizeof(float) * (size_t)ROWS * D_,
                       hipMemcpyDeviceToDevice, stream);

  // ---- retrieval ----
  float* outq = out + (size_t)ROWS * D_;                       // qwp region (B x 2D)
  pooled_kernel<<<B_, 256, 0, stream>>>(x, Wp, bp, pooled, pninv, outq);
  sim_kernel<<<N_ / 8, 256, 0, stream>>>(index_emb, pooled, pninv, sim);
  topk_kernel<<<B_, 256, 0, stream>>>(sim, seeds);
  zero_kernel<<<(B_ * N_ + 255) / 256, 256, 0, stream>>>(counts, B_ * N_);
  walk_kernel<<<(B_ * TOPW + 255) / 256, 256, 0, stream>>>(seeds, neighbors, walk_rand, counts);
  best_kernel<<<B_, 256, 0, stream>>>(sim, counts, index_emb, outq);
}
